// Time_Mix_63410897158761
// MI455X (gfx1250) — matrix operand order repaired and checked by reading, not compiled, not run on hardware
//
#include <hip/hip_runtime.h>
#include <math.h>

#define T_   512
#define E_   1024
#define H_   16
#define D_   64
#define HID_ 32

typedef __attribute__((ext_vector_type(8)))  __bf16 bf16x8;
typedef __attribute__((ext_vector_type(16))) __bf16 bf16x16;
typedef __attribute__((ext_vector_type(8)))  float  floatx8;

struct Ptr5 { const float* p[5]; };

enum { MODE_F32 = 0, MODE_TANH_BF16 = 1, MODE_SHIFT_BF16 = 2, MODE_DECAY_F32 = 3 };

// ---------------------------------------------------------------------------
// Weight transpose + f32->bf16 convert:  dst[b][c][r] = (bf16) src[b][r][c]
// ---------------------------------------------------------------------------
__global__ void transpose_bf16_kernel(__bf16* __restrict__ dst,
                                      const float* __restrict__ src,
                                      int batch, int rows, int cols)
{
    long idx = (long)blockIdx.x * blockDim.x + threadIdx.x;
    long total = (long)batch * rows * cols;
    if (idx >= total) return;
    int rc  = rows * cols;
    int b   = (int)(idx / rc);
    int rem = (int)(idx % rc);
    int r   = rem / cols;
    int c   = rem % cols;
    dst[(long)b * rc + (long)c * rows + r] = (__bf16)src[idx];
}

__global__ void add2_kernel(float* __restrict__ dst, const float* __restrict__ a,
                            const float* __restrict__ b, int n)
{
    int i = blockIdx.x * blockDim.x + threadIdx.x;
    if (i < n) dst[i] = a[i] + b[i];
}

// ---------------------------------------------------------------------------
// Token shift prep: diff, per-branch ddlerp input (bf16), gate mix (bf16)
// ---------------------------------------------------------------------------
__global__ void prep_kernel(const float* __restrict__ x, Ptr5 mus,
                            const float* __restrict__ gmu,
                            float* __restrict__ diff,
                            __bf16* __restrict__ XX, __bf16* __restrict__ XG)
{
    int idx = blockIdx.x * blockDim.x + threadIdx.x;
    if (idx >= T_ * E_) return;
    int t = idx >> 10;
    int e = idx & (E_ - 1);
    float xv = x[idx];
    float xl = (t > 0) ? x[idx - E_] : 0.f;
    float d  = xv - xl;
    diff[idx] = d;
#pragma unroll
    for (int b = 0; b < 5; ++b)
        XX[(long)b * (T_ * E_) + idx] = (__bf16)(xv + d * mus.p[b][e]);
    float gm = gmu[e];
    XG[idx] = (__bf16)(xv * gm + xl * (1.f - gm));
}

// ---------------------------------------------------------------------------
// 16x16 C-tile epilogue (D layout: VGPR r -> m = m0 + r + 8*half, n = n0+lm)
// ---------------------------------------------------------------------------
__device__ __forceinline__
void store_tile(void* __restrict__ Cv, int mode, long cbase, int ldc,
                int m0, int n, int half, floatx8 acc, float bn,
                const float* __restrict__ xf, const float* __restrict__ df)
{
#pragma unroll
    for (int r = 0; r < 8; ++r) {
        const int  mm = m0 + r + half * 8;
        const long ci = cbase + (long)mm * ldc + n;
        const float val = acc[r] + bn;
        if (mode == MODE_F32) {
            ((float*)Cv)[ci] = val;
        } else if (mode == MODE_TANH_BF16) {
            ((__bf16*)Cv)[ci] = (__bf16)tanhf(val);
        } else if (mode == MODE_SHIFT_BF16) {
            ((__bf16*)Cv)[ci] = (__bf16)(xf[ci] + val * df[ci]);  // x + lora*diff
        } else {                                                   // decay branch
            ((float*)Cv)[ci] = expf(-expf(val));
        }
    }
}

__device__ __forceinline__
bf16x16 frag16(bf16x8 lo, bf16x8 hi)
{
    return __builtin_shufflevector(lo, hi, 0,1,2,3,4,5,6,7,8,9,10,11,12,13,14,15);
}

// ---------------------------------------------------------------------------
// Wave-per-32x32-tile bf16 WMMA GEMM:  C[M,N] = A[M,K] * BT[N,K]^T + bias[N]
// 2x2 register blocking: per K-step, 8 b128 loads feed 4 independent
// v_wmma_f32_16x16x32_bf16 chains (2 loads/WMMA; A and B fragments each
// reused twice). Fragment loads are contiguous b128s because BT is
// pre-transposed (ISA 7.12.2 layouts: A lane(l): m=l%16, K chunks at
// k0+half*8 and k0+16+half*8 ; B lane(l): n=l%16, K chunks at k0+half*8 and k0+16+half*8).
// Batched over blockIdx.y (per-head dense) via element strides.
// ---------------------------------------------------------------------------
__global__ __launch_bounds__(256)
void wmma_gemm_kernel(const __bf16* __restrict__ A,  int lda, long sA,
                      const __bf16* __restrict__ BT, int ldb, long sB,
                      const float* __restrict__ bias, long sBias,
                      void* __restrict__ Cv, int ldc, long sC,
                      int M, int N, int K, int mode,
                      const float* __restrict__ xf, const float* __restrict__ df)
{
    const int z    = blockIdx.y;
    const int wave = threadIdx.x >> 5;
    const int lane = threadIdx.x & 31;
    const int Mt = M >> 5, Nt = N >> 5;               // 32x32 macro-tiles
    const int tile = blockIdx.x * 8 + wave;
    if (tile >= Mt * Nt) return;                      // uniform per-wave: EXEC stays all-1s

    const __bf16* Az = A  + (long)z * sA;
    const __bf16* Bz = BT + (long)z * sB;
    const float*  bz = bias + (long)z * sBias;

    const int m0   = (tile / Nt) << 5;
    const int n0   = (tile % Nt) << 5;
    const int half = lane >> 4;
    const int lm   = lane & 15;

    const __bf16* arow0 = Az + (long)(m0 + lm) * lda + half * 8;        // rows m0..m0+15
    const __bf16* arow1 = arow0 + (long)16 * lda;                       // rows m0+16..+31
    const __bf16* brow0 = Bz + (long)(n0 + lm) * ldb + half * 8;        // cols n0..n0+15
    const __bf16* brow1 = brow0 + (long)16 * ldb;                       // cols n0+16..+31

    floatx8 acc00 = {0.f,0.f,0.f,0.f,0.f,0.f,0.f,0.f};
    floatx8 acc01 = acc00, acc10 = acc00, acc11 = acc00;

    for (int k0 = 0; k0 < K; k0 += 32) {
        bf16x8 a0lo = *(const bf16x8*)(arow0 + k0);
        bf16x8 a0hi = *(const bf16x8*)(arow0 + k0 + 16);
        bf16x8 a1lo = *(const bf16x8*)(arow1 + k0);
        bf16x8 a1hi = *(const bf16x8*)(arow1 + k0 + 16);
        bf16x8 b0lo = *(const bf16x8*)(brow0 + k0);
        bf16x8 b0hi = *(const bf16x8*)(brow0 + k0 + 16);
        bf16x8 b1lo = *(const bf16x8*)(brow1 + k0);
        bf16x8 b1hi = *(const bf16x8*)(brow1 + k0 + 16);
        if (k0 + 32 < K) {                            // near-scope prefetch of next K-slab
            __builtin_prefetch((const void*)(arow0 + k0 + 32), 0, 3);
            __builtin_prefetch((const void*)(arow1 + k0 + 32), 0, 3);
            __builtin_prefetch((const void*)(brow0 + k0 + 32), 0, 3);
            __builtin_prefetch((const void*)(brow1 + k0 + 32), 0, 3);
        }
        bf16x16 a0 = frag16(a0lo, a0hi);
        bf16x16 a1 = frag16(a1lo, a1hi);
        bf16x16 b0 = frag16(b0lo, b0hi);
        bf16x16 b1 = frag16(b1lo, b1hi);
        acc00 = __builtin_amdgcn_wmma_f32_16x16x32_bf16(false, a0, false, b0, (short)0, acc00, false, false);
        acc01 = __builtin_amdgcn_wmma_f32_16x16x32_bf16(false, a0, false, b1, (short)0, acc01, false, false);
        acc10 = __builtin_amdgcn_wmma_f32_16x16x32_bf16(false, a1, false, b0, (short)0, acc10, false, false);
        acc11 = __builtin_amdgcn_wmma_f32_16x16x32_bf16(false, a1, false, b1, (short)0, acc11, false, false);
    }

    const long cbase = (long)z * sC;
    const float bn0 = bz[n0 + lm];
    const float bn1 = bz[n0 + 16 + lm];
    store_tile(Cv, mode, cbase, ldc, m0,      n0 + lm,      half, acc00, bn0, xf, df);
    store_tile(Cv, mode, cbase, ldc, m0,      n0 + 16 + lm, half, acc01, bn1, xf, df);
    store_tile(Cv, mode, cbase, ldc, m0 + 16, n0 + lm,      half, acc10, bn0, xf, df);
    store_tile(Cv, mode, cbase, ldc, m0 + 16, n0 + 16 + lm, half, acc11, bn1, xf, df);
}

// ---------------------------------------------------------------------------
// Per-head linear recurrence (the only serial part).
// Block = 64 threads (2 waves) per head; thread i owns state row S[i][0..63]
// in VGPRs. Per step: rwkv_i = u_i k_i (v.r) + sum_j S[i][j] r_j ; then
// S[i][j] = w_i S[i][j] + k_i v_j.
// ---------------------------------------------------------------------------
__global__ __launch_bounds__(64)
void scan_kernel(const float* __restrict__ kb, const float* __restrict__ rb,
                 const float* __restrict__ vb, const float* __restrict__ ub,
                 const float* __restrict__ wb, float* __restrict__ out)
{
    const int h = blockIdx.x;
    const int i = threadIdx.x;
    __shared__ float vs[D_], rs[D_];
    float S[D_];
#pragma unroll
    for (int j = 0; j < D_; ++j) S[j] = 0.f;

    for (int t = 0; t < T_; ++t) {
        const int base = t * E_ + h * D_;
        const float ki = kb[base + i];
        const float wi = wb[base + i];
        const float ui = ub[base + i];
        vs[i] = vb[base + i];
        rs[i] = rb[base + i];
        __syncthreads();
        float dot = 0.f, vr = 0.f;
#pragma unroll
        for (int j = 0; j < D_; ++j) {
            dot = fmaf(S[j], rs[j], dot);
            vr  = fmaf(vs[j], rs[j], vr);
        }
        out[base + i] = ui * ki * vr + dot;       // uses S_{t-1} (pre-update)
#pragma unroll
        for (int j = 0; j < D_; ++j) S[j] = fmaf(wi, S[j], ki * vs[j]);
        __syncthreads();
    }
}

// ---------------------------------------------------------------------------
// GroupNorm over each head's 64 channels + silu(gate), output as bf16 GEMM A.
// ---------------------------------------------------------------------------
__global__ __launch_bounds__(64)
void gn_gate_kernel(const float* __restrict__ rwkv, const float* __restrict__ g,
                    const float* __restrict__ gamma, const float* __restrict__ beta,
                    __bf16* __restrict__ y)
{
    const int t = blockIdx.x, h = blockIdx.y, i = threadIdx.x;
    __shared__ float red[D_];
    const int  e   = h * D_ + i;
    const long idx = (long)t * E_ + e;
    float v = rwkv[idx];
    red[i] = v; __syncthreads();
    for (int s = 32; s > 0; s >>= 1) { if (i < s) red[i] += red[i + s]; __syncthreads(); }
    float mean = red[0] * (1.f / D_); __syncthreads();
    float d = v - mean;
    red[i] = d * d; __syncthreads();
    for (int s = 32; s > 0; s >>= 1) { if (i < s) red[i] += red[i + s]; __syncthreads(); }
    float var = red[0] * (1.f / D_);
    float nv = d * rsqrtf(var + 1e-3f) * gamma[e] + beta[e];
    float gv = g[idx];
    float sg = gv / (1.f + expf(-gv));            // silu
    y[idx] = (__bf16)(nv * sg);
}

// ---------------------------------------------------------------------------
// Host-side orchestration (graph-capture safe: only kernel launches on stream)
// ---------------------------------------------------------------------------
extern "C" void kernel_launch(void* const* d_in, const int* in_sizes, int n_in,
                              void* d_out, int out_size, void* d_ws, size_t ws_size,
                              hipStream_t stream)
{
    (void)in_sizes; (void)n_in; (void)out_size; (void)ws_size;

    // setup_inputs() dict insertion order:
    // [0]=inputs, branches key/receptance/decay/value/bonus each
    // (mu,lam,Aw,Ab,Bw,Bb,W,b) at [1+8b .. 8+8b], then
    // [41]=gate_mu [42]=gate_W [43]=gate_b [44]=gn_gamma [45]=gn_beta
    // [46]=out_W [47]=out_b
    const float* x = (const float*)d_in[0];
    auto br = [&](int b, int item) -> const float* { return (const float*)d_in[1 + 8 * b + item]; };
    const float* gate_mu = (const float*)d_in[41];
    const float* gate_W  = (const float*)d_in[42];
    const float* gate_b  = (const float*)d_in[43];
    const float* gn_g    = (const float*)d_in[44];
    const float* gn_b    = (const float*)d_in[45];
    const float* out_W   = (const float*)d_in[46];
    const float* out_b   = (const float*)d_in[47];

    char* ws = (char*)d_ws;
    const size_t TE = (size_t)T_ * E_;
    size_t off = 0;
    auto take = [&](size_t bytes) { size_t o = off; off += (bytes + 255) & ~(size_t)255; return o; };
    float*  diff  = (float*)(ws + take(TE * 4));
    __bf16* XX    = (__bf16*)(ws + take(5 * TE * 2));
    __bf16* XG    = (__bf16*)(ws + take(TE * 2));
    __bf16* AwT   = (__bf16*)(ws + take(5 * (size_t)E_ * HID_ * 2));
    __bf16* BwT   = (__bf16*)(ws + take(5 * (size_t)E_ * HID_ * 2));
    __bf16* WT    = (__bf16*)(ws + take(5 * (size_t)H_ * D_ * D_ * 2));
    __bf16* gWT   = (__bf16*)(ws + take((size_t)E_ * E_ * 2));
    __bf16* oWT   = (__bf16*)(ws + take((size_t)E_ * E_ * 2));
    float*  bias2 = (float*)(ws + take(5 * (size_t)E_ * 4));
    __bf16* H1    = (__bf16*)(ws + take(5 * (size_t)T_ * HID_ * 2));
    __bf16* SH    = (__bf16*)(ws + take(5 * TE * 2));
    float*  kbuf  = (float*)(ws + take(TE * 4));
    float*  rbuf  = (float*)(ws + take(TE * 4));
    float*  wbuf  = (float*)(ws + take(TE * 4));
    float*  vbuf  = (float*)(ws + take(TE * 4));
    float*  ubuf  = (float*)(ws + take(TE * 4));
    float*  gbuf  = (float*)(ws + take(TE * 4));
    float*  rwbuf = (float*)(ws + take(TE * 4));
    __bf16* ybf   = (__bf16*)(ws + take(TE * 2));

    // 1) weight transposes -> bf16 N^T x K, plus combined (Bb + lam) bias
    auto tr = [&](__bf16* dst, const float* src, int batch, int rows, int cols) {
        long total = (long)batch * rows * cols;
        transpose_bf16_kernel<<<(int)((total + 255) / 256), 256, 0, stream>>>(dst, src, batch, rows, cols);
    };
    tr(gWT, gate_W, 1, E_, E_);
    tr(oWT, out_W, 1, E_, E_);
    for (int b = 0; b < 5; ++b) {
        tr(AwT + (size_t)b * E_ * HID_, br(b, 2), 1, E_, HID_);     // Aw (E,HID) -> (HID,E)
        tr(BwT + (size_t)b * E_ * HID_, br(b, 4), 1, HID_, E_);     // Bw (HID,E) -> (E,HID)
        tr(WT  + (size_t)b * H_ * D_ * D_, br(b, 6), H_, D_, D_);   // W  (H,D,D) -> (H,D^T)
        add2_kernel<<<4, 256, 0, stream>>>(bias2 + b * E_, br(b, 5), br(b, 1), E_);
    }

    // 2) token-shift prep
    Ptr5 mus;
    for (int b = 0; b < 5; ++b) mus.p[b] = br(b, 0);
    prep_kernel<<<(T_ * E_ + 255) / 256, 256, 0, stream>>>(x, mus, gate_mu, diff, XX, XG);

    auto gemm = [&](const __bf16* A, int lda, long sA,
                    const __bf16* BT, int ldb, long sB,
                    const float* bias, long sBias,
                    void* C, int ldc, long sC,
                    int M, int N, int K, int mode, int batch,
                    const float* xf, const float* df) {
        int tiles = (M >> 5) * (N >> 5);              // 32x32 macro-tiles, 8 waves/block
        dim3 grid((tiles + 7) / 8, batch);
        wmma_gemm_kernel<<<grid, 256, 0, stream>>>(A, lda, sA, BT, ldb, sB, bias, sBias,
                                                   C, ldc, sC, M, N, K, mode, xf, df);
    };

    // branch outputs: 0=key->k 1=receptance->r 2=decay->w 3=value->v 4=bonus->u
    float* headOut[5] = { kbuf, rbuf, wbuf, vbuf, ubuf };
    for (int b = 0; b < 5; ++b) {
        // lora down: H1 = tanh(XX @ Aw + Ab)            [512x32x1024]
        gemm(XX + (size_t)b * TE, E_, 0,
             AwT + (size_t)b * E_ * HID_, E_, 0,
             br(b, 3), 0,
             H1 + (size_t)b * T_ * HID_, HID_, 0,
             T_, HID_, E_, MODE_TANH_BF16, 1, nullptr, nullptr);
        // lora up (+ fused ddlerp): SH = x + (H1 @ Bw + Bb + lam) * diff   [512x1024x32]
        gemm(H1 + (size_t)b * T_ * HID_, HID_, 0,
             BwT + (size_t)b * E_ * HID_, HID_, 0,
             bias2 + b * E_, 0,
             SH + (size_t)b * TE, E_, 0,
             T_, E_, HID_, MODE_SHIFT_BF16, 1, x, diff);
        // per-head dense, batched over heads on blockIdx.y  [16 x (512x64x64)]
        gemm(SH + (size_t)b * TE, E_, D_,
             WT + (size_t)b * H_ * D_ * D_, D_, D_ * D_,
             br(b, 7), D_,
             headOut[b], E_, D_,
             T_, D_, D_, (b == 2) ? MODE_DECAY_F32 : MODE_F32, H_, nullptr, nullptr);
    }

    // gate: g = XG @ gate_W + gate_b                     [512x1024x1024]
    gemm(XG, E_, 0, gWT, E_, 0, gate_b, 0, gbuf, E_, 0, T_, E_, E_, MODE_F32, 1, nullptr, nullptr);

    // sequential recurrence per head
    scan_kernel<<<H_, D_, 0, stream>>>(kbuf, rbuf, vbuf, ubuf, wbuf, rwbuf);

    // group norm + silu(gate) -> bf16
    gn_gate_kernel<<<dim3(T_, H_), D_, 0, stream>>>(rwbuf, gbuf, gn_g, gn_b, ybf);

    // output projection -> f32 d_out                     [512x1024x1024]
    gemm(ybf, E_, 0, oWT, E_, 0, out_b, 0, (float*)d_out, E_, 0, T_, E_, E_, MODE_F32, 1, nullptr, nullptr);
}
